// AttentionHead_85925115724468
// MI455X (gfx1250) — matrix operand order repaired and checked by reading, not compiled, not run on hardware
//
#include <hip/hip_runtime.h>

typedef _Float16 h16 __attribute__((ext_vector_type(16)));
typedef _Float16 h2  __attribute__((ext_vector_type(2)));
typedef float    f8  __attribute__((ext_vector_type(8)));
typedef int      v4i __attribute__((vector_size(16)));   // matches async builtin param type

#define EMBED 1024
#define HS    64
#define SEQ   4096
#define BATCH 8

// ---- fragment index helpers (CDNA5 16-bit WMMA VGPR layouts, ISA 7.12.2) ----
// A (16x32, MxK): lanes 0-15 M=lane, V0:K=0,1 .. V3:K=6,7, V4:K=16,17 .. V7:K=22,23
//                 lanes 16-31: same M, K offset +8
__device__ __forceinline__ int a_kbase(int p, int lane) {
    int k = (p < 4) ? (2 * p) : (2 * p + 8);
    return k + ((lane < 16) ? 0 : 8);
}
// B (32x16, KxN): lanes 0-15 N=lane, V0:K=0,1 .. V3:K=6,7, V4:K=16,17 .. V7:K=22,23; lanes 16-31: K offset +8
__device__ __forceinline__ int b_kbase(int p, int lane) {
    int k = (p < 4) ? (2 * p) : (2 * p + 8);
    return k + ((lane < 16) ? 0 : 8);
}

// ---- gfx1250 async global->LDS copy (ASYNCcnt-tracked), builtin or asm ----
__device__ __forceinline__ void async_copy_b128(const void* g, void* l) {
#if __has_builtin(__builtin_amdgcn_global_load_async_to_lds_b128)
    __builtin_amdgcn_global_load_async_to_lds_b128((v4i*)g, (v4i*)l, 0, 0);
#else
    unsigned lo = (unsigned)(unsigned long long)(uintptr_t)l;  // LDS offset
    asm volatile("global_load_async_to_lds_b128 %0, %1, off"
                 :: "v"(lo), "v"((unsigned long long)(uintptr_t)g) : "memory");
#endif
}
__device__ __forceinline__ void wait_async0() {
#if __has_builtin(__builtin_amdgcn_s_wait_asynccnt)
    __builtin_amdgcn_s_wait_asynccnt(0);
#else
    asm volatile("s_wait_asynccnt 0x0" ::: "memory");
#endif
}

// =====================================================================
// Kernel 0: convert W [1024][64] f32 -> Wt [64][1024] f16 (x3 matrices)
// so projection B-fragments read k-contiguous halfs of one Wt row (h2 loads).
// =====================================================================
__global__ __launch_bounds__(256)
void wt_convert_kernel(const float* __restrict__ Wk,
                       const float* __restrict__ Wq,
                       const float* __restrict__ Wv,
                       _Float16* __restrict__ wtk,
                       _Float16* __restrict__ wtq,
                       _Float16* __restrict__ wtv)
{
    int idx = blockIdx.x * 256 + threadIdx.x;   // 0 .. 64K-1
    int h = idx & (HS - 1);
    int e = idx >> 6;
    long t = (long)h * EMBED + e;
    wtk[t] = (_Float16)Wk[idx];
    wtq[t] = (_Float16)Wq[idx];
    wtv[t] = (_Float16)Wv[idx];
}

// =====================================================================
// Kernel 1: fused QKV projection.  Block = 128 thr (4 waves); wave w owns
// output n-tile w.  x chunk (16x32 f32, 2KB) staged by double-buffered
// async-to-LDS; converted to f16 during A-fragment build (shared by 3 WMMAs).
// Outputs: q,k as [b*seq][64] f16 ; v transposed as [b][64][seq] f16.
// =====================================================================
__global__ __launch_bounds__(128)
void qkv_proj_kernel(const float* __restrict__ x,
                     const _Float16* __restrict__ wtk,
                     const _Float16* __restrict__ wtq,
                     const _Float16* __restrict__ wtv,
                     _Float16* __restrict__ qh,
                     _Float16* __restrict__ kh,
                     _Float16* __restrict__ vTh)
{
    __shared__ __align__(16) float xlds[2][16 * 32];
    const int tid  = threadIdx.x;
    const int lane = tid & 31;
    const int wave = tid >> 5;                  // n-tile 0..3
    const long rowbase = (long)blockIdx.x * 16; // global row = b*SEQ + seq
    const int n = lane & 15;
    const int ncol = wave * 16 + n;

    // async stage: 128 threads x 16B = 2KB = 16 rows x 32 f32
    const int xr = tid >> 3;            // row 0..15
    const int xc = (tid & 7) * 4;       // col 0..28 step 4
    const float* xbase = x + (rowbase + xr) * EMBED + xc;

    f8 cq = {}; f8 ck = {}; f8 cv = {};

    async_copy_b128(xbase, &xlds[0][xr * 32 + xc]);
    wait_async0();
    __syncthreads();

    const int NCHUNK = EMBED / 32;
    for (int i = 0; i < NCHUNK; ++i) {
        const int buf = i & 1;
        const int kc  = i * 32;
        if (i + 1 < NCHUNK)
            async_copy_b128(xbase + kc + 32, &xlds[buf ^ 1][xr * 32 + xc]);

        // A fragment: f32 pairs from LDS, convert to f16 (shared by q/k/v)
        h16 a;
        #pragma unroll
        for (int p = 0; p < 8; ++p) {
            int kb = a_kbase(p, lane);
            float2 pv = *(const float2*)(&xlds[buf][(lane & 15) * 32 + kb]);
            a[2 * p] = (_Float16)pv.x; a[2 * p + 1] = (_Float16)pv.y;
        }
        // B fragments: Wt[ncol][kc + kk], two runs of 8 contiguous halfs per lane
        h16 bq, bk, bv;
        #pragma unroll
        for (int p = 0; p < 8; ++p) {
            int kk = b_kbase(p, lane);
            h2 q2 = *(const h2*)(wtq + (long)ncol * EMBED + kc + kk);
            h2 k2 = *(const h2*)(wtk + (long)ncol * EMBED + kc + kk);
            h2 v2 = *(const h2*)(wtv + (long)ncol * EMBED + kc + kk);
            bq[2 * p] = q2[0]; bq[2 * p + 1] = q2[1];
            bk[2 * p] = k2[0]; bk[2 * p + 1] = k2[1];
            bv[2 * p] = v2[0]; bv[2 * p + 1] = v2[1];
        }
        cq = __builtin_amdgcn_wmma_f32_16x16x32_f16(false, a, false, bq, (short)0, cq, false, false);
        ck = __builtin_amdgcn_wmma_f32_16x16x32_f16(false, a, false, bk, (short)0, ck, false, false);
        cv = __builtin_amdgcn_wmma_f32_16x16x32_f16(false, a, false, bv, (short)0, cv, false, false);

        if (i + 1 < NCHUNK) wait_async0();
        __syncthreads();
    }

    // C layout: lane<16 -> M=r, else M=r+8; N = lane&15
    #pragma unroll
    for (int r = 0; r < 8; ++r) {
        int row = (lane < 16) ? r : (r + 8);
        long grow = rowbase + row;
        long off  = grow * HS + ncol;
        qh[off] = (_Float16)cq[r];
        kh[off] = (_Float16)ck[r];
        int b   = (int)(grow / SEQ);
        int seq = (int)(grow % SEQ);
        vTh[((long)b * HS + ncol) * SEQ + seq] = (_Float16)cv[r];
    }
}

// =====================================================================
// Kernel 2: flash attention.  Block = 256 thr (8 waves); all waves share
// the same 32-key chunk, so K (32x64) and V^T (64x32) tiles are staged in
// LDS via double-buffered async-to-LDS (one b128 per thread per tile).
// One wave per 16-query tile, online softmax, WMMA for S and PV.
// =====================================================================
__global__ __launch_bounds__(256)
void flash_attn_kernel(const _Float16* __restrict__ qh,
                       const _Float16* __restrict__ kh,
                       const _Float16* __restrict__ vTh,
                       float* __restrict__ out)
{
    __shared__ __align__(16) _Float16 Klds[2][32 * 64];  // [key][d]  4KB each
    __shared__ __align__(16) _Float16 Vlds[2][64 * 32];  // [hd][key] 4KB each
    __shared__ __align__(16) _Float16 Plds[8][16 * 32];  // per-wave P staging
    const int tid   = threadIdx.x;
    const int lane  = tid & 31;
    const int wave  = tid >> 5;
    const int b     = blockIdx.y;
    const int qbase = blockIdx.x * 128 + wave * 16;

    const _Float16* qptr = qh  + (long)b * SEQ * HS;
    const _Float16* kptr = kh  + (long)b * SEQ * HS;
    const _Float16* vptr = vTh + (long)b * HS * SEQ;
    _Float16* pl = &Plds[wave][0];

    // per-thread async-copy slots
    const int khalf = tid * 8;                 // K tile: contiguous 4KB
    const int vhd   = tid >> 2;                // V tile: 64 rows x 64B
    const int vpc   = (tid & 3) * 8;

    // Q: 16x64 -> two A fragments (d = 0..31, 32..63)
    h16 qa0, qa1;
    {
        int m = lane & 15;
        #pragma unroll
        for (int p = 0; p < 8; ++p) {
            int kb = a_kbase(p, lane);
            h2 v0 = *(const h2*)(qptr + (long)(qbase + m) * HS + kb);
            h2 v1 = *(const h2*)(qptr + (long)(qbase + m) * HS + 32 + kb);
            qa0[2 * p] = v0[0]; qa0[2 * p + 1] = v0[1];
            qa1[2 * p] = v1[0]; qa1[2 * p + 1] = v1[1];
        }
    }

    f8 o0 = {}, o1 = {}, o2 = {}, o3 = {};
    float M[8], L[8];
    #pragma unroll
    for (int r = 0; r < 8; ++r) { M[r] = -3.0e38f; L[r] = 0.0f; }

    // prologue: stage chunk 0 into buffer 0
    async_copy_b128(kptr + khalf, &Klds[0][khalf]);
    async_copy_b128(vptr + (long)vhd * SEQ + vpc, &Vlds[0][vhd * 32 + vpc]);
    wait_async0();
    __syncthreads();

    const int NIT = SEQ / 32;
    for (int it = 0; it < NIT; ++it) {
        const int buf = it & 1;
        const int kb0 = it * 32;
        if (it + 1 < NIT) {   // issue next chunk into the other buffer
            async_copy_b128(kptr + (long)(kb0 + 32) * HS + khalf, &Klds[buf ^ 1][khalf]);
            async_copy_b128(vptr + (long)vhd * SEQ + kb0 + 32 + vpc, &Vlds[buf ^ 1][vhd * 32 + vpc]);
        }

        // ---- S = Q * K^T : two 16-key C tiles, K-dim 64 in two chunks ----
        f8 s0 = {}, s1 = {};
        {
            int n = lane & 15;
            #pragma unroll
            for (int dchunk = 0; dchunk < 2; ++dchunk) {
                h16 bb0, bb1;
                #pragma unroll
                for (int p = 0; p < 8; ++p) {
                    int kk = b_kbase(p, lane);  // head-dim index within chunk
                    h2 v0 = *(const h2*)(&Klds[buf][n * 64 + dchunk * 32 + kk]);
                    h2 v1 = *(const h2*)(&Klds[buf][(16 + n) * 64 + dchunk * 32 + kk]);
                    bb0[2 * p] = v0[0]; bb0[2 * p + 1] = v0[1];
                    bb1[2 * p] = v1[0]; bb1[2 * p + 1] = v1[1];
                }
                const h16 qa = dchunk ? qa1 : qa0;
                s0 = __builtin_amdgcn_wmma_f32_16x16x32_f16(false, qa, false, bb0, (short)0, s0, false, false);
                s1 = __builtin_amdgcn_wmma_f32_16x16x32_f16(false, qa, false, bb1, (short)0, s1, false, false);
            }
        }
        // ---- online softmax (row = C-fragment row mapping; 16-lane reduce) ----
        #pragma unroll
        for (int r = 0; r < 8; ++r) {
            float a0 = s0[r] * 0.125f;   // 1/sqrt(64)
            float a1 = s1[r] * 0.125f;
            float mx = fmaxf(a0, a1);
            #pragma unroll
            for (int off = 8; off >= 1; off >>= 1) mx = fmaxf(mx, __shfl_xor(mx, off, 16));
            float mnew  = fmaxf(M[r], mx);
            float alpha = __expf(M[r] - mnew);
            float p0 = __expf(a0 - mnew);
            float p1 = __expf(a1 - mnew);
            float rs = p0 + p1;
            #pragma unroll
            for (int off = 8; off >= 1; off >>= 1) rs += __shfl_xor(rs, off, 16);
            L[r] = L[r] * alpha + rs;
            M[r] = mnew;
            o0[r] *= alpha; o1[r] *= alpha; o2[r] *= alpha; o3[r] *= alpha;
            int row = (lane < 16) ? r : (r + 8);
            int col = lane & 15;
            pl[row * 32 + col]      = (_Float16)p0;
            pl[row * 32 + col + 16] = (_Float16)p1;
        }
        __builtin_amdgcn_wave_barrier();
        asm volatile("s_wait_dscnt 0x0" ::: "memory");

        // ---- P (C layout) -> A fragment via per-wave LDS ----
        h16 pa;
        {
            int m = lane & 15;
            #pragma unroll
            for (int p = 0; p < 8; ++p) {
                h2 v = *(const h2*)(pl + m * 32 + a_kbase(p, lane));
                pa[2 * p] = v[0]; pa[2 * p + 1] = v[1];
            }
        }
        __builtin_amdgcn_wave_barrier();

        // ---- O += P * V : 4 head-dim n-tiles from LDS V^T tile ----
        {
            int n = lane & 15;
            #pragma unroll
            for (int t = 0; t < 4; ++t) {
                h16 vb;
                #pragma unroll
                for (int p = 0; p < 8; ++p) {
                    int kk = b_kbase(p, lane);  // key index within chunk
                    h2 v = *(const h2*)(&Vlds[buf][(t * 16 + n) * 32 + kk]);
                    vb[2 * p] = v[0]; vb[2 * p + 1] = v[1];
                }
                f8& o = (t == 0) ? o0 : (t == 1) ? o1 : (t == 2) ? o2 : o3;
                o = __builtin_amdgcn_wmma_f32_16x16x32_f16(false, pa, false, vb, (short)0, o, false, false);
            }
        }

        if (it + 1 < NIT) wait_async0();
        __syncthreads();   // publish next buffer; retire this one for overwrite
    }

    // ---- normalize and write out [b][q][64] f32 ----
    int col = lane & 15;
    #pragma unroll
    for (int r = 0; r < 8; ++r) {
        int row = (lane < 16) ? r : (r + 8);
        float inv = 1.0f / L[r];
        long base = ((long)b * SEQ + qbase + row) * HS;
        out[base + col]      = o0[r] * inv;
        out[base + 16 + col] = o1[r] * inv;
        out[base + 32 + col] = o2[r] * inv;
        out[base + 48 + col] = o3[r] * inv;
    }
}

extern "C" void kernel_launch(void* const* d_in, const int* in_sizes, int n_in,
                              void* d_out, int out_size, void* d_ws, size_t ws_size,
                              hipStream_t stream) {
    const float* x  = (const float*)d_in[0];
    const float* Wk = (const float*)d_in[1];
    const float* Wq = (const float*)d_in[2];
    const float* Wv = (const float*)d_in[3];
    float* out = (float*)d_out;

    // workspace: q (4MB) | k (4MB) | v^T (4MB) | Wt f16 x3 (128KB each)
    _Float16* qh  = (_Float16*)d_ws;
    _Float16* kh  = qh  + (size_t)BATCH * SEQ * HS;
    _Float16* vTh = kh  + (size_t)BATCH * SEQ * HS;
    _Float16* wtk = vTh + (size_t)BATCH * SEQ * HS;
    _Float16* wtq = wtk + (size_t)HS * EMBED;
    _Float16* wtv = wtq + (size_t)HS * EMBED;

    wt_convert_kernel<<<dim3(EMBED * HS / 256), 256, 0, stream>>>(Wk, Wq, Wv, wtk, wtq, wtv);

    dim3 g1(BATCH * SEQ / 16);          // 2048 row tiles
    qkv_proj_kernel<<<g1, 128, 0, stream>>>(x, wtk, wtq, wtv, qh, kh, vTh);

    dim3 g2(SEQ / 128, BATCH);          // 32 x 8 blocks, 8 waves each
    flash_attn_kernel<<<g2, 256, 0, stream>>>(qh, kh, vTh, out);
}
